// ResMixerBlock_78323023610392
// MI455X (gfx1250) — compile-verified, matrix operand order checked by reading, not run on hardware
//
#include <hip/hip_runtime.h>

// ---------------------------------------------------------------------------
// ResMixerBlock for MI455X (gfx1250): 64 sequential steps, each =
//   value net (4x conv1x1 GEMMs over N=512 cols) + gate net (4x GEMMs over
//   N=128 cols) + GN/GELU elementwise stages + exp-gate combine.
// GEMMs: v_wmma_f32_16x16x32_f16, 64x128 block tile, 32x32 per wave (4 WMMA
// per K-step), double-buffered LDS panels filled by global_load_async_to_lds
// (ASYNCcnt-tracked, bypasses VGPRs), GN stats in f32.
// ---------------------------------------------------------------------------

typedef __attribute__((ext_vector_type(16))) _Float16 v16h;
typedef __attribute__((ext_vector_type(8)))  float    v8f;

#define EPSF 1e-5f

__device__ __forceinline__ float gelu_f(float x) {
    return 0.5f * x * (1.0f + erff(x * 0.7071067811865475f));
}

struct Halves32 { uint4 lo, hi; };   // 32 bytes == v16h

__device__ __forceinline__ v16h load_frag(const _Float16* p) {
    Halves32 h;
    h.lo = *reinterpret_cast<const uint4*>(p);
    h.hi = *reinterpret_cast<const uint4*>(p + 16);
    return __builtin_bit_cast(v16h, h);
}

// Async Global->LDS 16B copy (gfx1250 GLOBAL_LOAD_ASYNC_TO_LDS_B128).
// lds: 32-bit LDS byte offset (low 32 bits of a generic LDS pointer),
// g:   64-bit global address.  Tracked by ASYNCcnt.
__device__ __forceinline__ void async_cp16(unsigned int lds, const void* g) {
    asm volatile("global_load_async_to_lds_b128 %0, %1, off"
                 :: "v"(lds), "v"((unsigned long long)(size_t)g) : "memory");
}
__device__ __forceinline__ void wait_async3() {
    asm volatile("s_wait_asynccnt 0x3" ::: "memory");
}
__device__ __forceinline__ void wait_async0() {
    asm volatile("s_wait_asynccnt 0x0" ::: "memory");
}

// ---------------------------------------------------------------------------
// Generic (optionally grouped) GEMM: Y[g*Mg+m, n] = sum_k W[g*Mg+m, k] *
// X[n, g*Kg + k] + bias.  W: f16 row-major [groups*Mg][Kg].  X: f16 with
// contiguous columns of length KX.
// Block: 256 threads = 8 waves (2 along M x 4 along N); block tile 64 x 128,
// wave tile 32 x 32 (2x2 WMMA subtiles).  K stepped by 32, double-buffered
// LDS panels (A 4KB, B 8KB per buffer) filled asynchronously.
// ---------------------------------------------------------------------------
__global__ __launch_bounds__(256)
void gemm_f16_wmma(const _Float16* __restrict__ W, const _Float16* __restrict__ X,
                   const float* __restrict__ bias, float* __restrict__ Y,
                   int Mg, int N, int Kg, int KX) {
    __shared__ alignas(16) _Float16 Al[2][64 * 32];
    __shared__ alignas(16) _Float16 Bl[2][128 * 32];

    const int g    = blockIdx.z;
    const int m0   = blockIdx.x * 64;
    const int n0   = blockIdx.y * 128;
    const int tid  = threadIdx.x;
    const int lane = tid & 31;
    const int wave = tid >> 5;
    const int wm   = wave >> 2;  // 2 waves along M (32 rows each)
    const int wn   = wave & 3;   // 4 waves along N (32 cols each)

    const _Float16* Wg = W + (size_t)g * Mg * Kg;
    const int kx0 = g * Kg;

    // Per-thread async staging coordinates.
    const int arow = tid >> 2, aseg = (tid & 3) * 8;   // A: 64 rows, 16B/thread
    const int bn   = tid >> 2, bseg = (tid & 3) * 8;   // B: cols bn and bn+64, 16B each

    const unsigned int ldsA0 = (unsigned int)(size_t)&Al[0][arow * 32 + aseg];
    const unsigned int ldsA1 = (unsigned int)(size_t)&Al[1][arow * 32 + aseg];
    const unsigned int ldsB0a = (unsigned int)(size_t)&Bl[0][bn * 32 + bseg];
    const unsigned int ldsB0b = (unsigned int)(size_t)&Bl[0][(bn + 64) * 32 + bseg];
    const unsigned int ldsB1a = (unsigned int)(size_t)&Bl[1][bn * 32 + bseg];
    const unsigned int ldsB1b = (unsigned int)(size_t)&Bl[1][(bn + 64) * 32 + bseg];

    auto issue = [&](int k0, int buf) {
        const _Float16* ga  = &Wg[(size_t)(m0 + arow) * Kg + k0 + aseg];
        const _Float16* gb0 = &X[(size_t)(n0 + bn) * KX + kx0 + k0 + bseg];
        const _Float16* gb1 = &X[(size_t)(n0 + bn + 64) * KX + kx0 + k0 + bseg];
        if (buf == 0) {
            async_cp16(ldsA0, ga);
            async_cp16(ldsB0a, gb0);
            async_cp16(ldsB0b, gb1);
        } else {
            async_cp16(ldsA1, ga);
            async_cp16(ldsB1a, gb0);
            async_cp16(ldsB1b, gb1);
        }
    };

    v8f acc00 = {}, acc01 = {}, acc10 = {}, acc11 = {};
    const int npan = Kg / 32;
    issue(0, 0);

    for (int p = 0; p < npan; ++p) {
        const int cur = p & 1;
        if (p + 1 < npan) {             // stream next panel into the other buffer
            issue((p + 1) * 32, cur ^ 1);
            wait_async3();              // loads complete in order: panel p landed
        } else {
            wait_async0();
        }
        __syncthreads();                // panel p visible to all waves

        // CDNA5 16-bit A 16x32 per-lane layout: lane<16 holds K {0..7,16..23},
        // lane>=16 holds K {8..15,24..31} of row m = lane&15.
        const int kb = (lane >> 4) * 8;
        const int mr = wm * 32 + (lane & 15);
        const int nc = wn * 32 + (lane & 15);
        v16h a0 = load_frag(&Al[cur][mr * 32 + kb]);
        v16h a1 = load_frag(&Al[cur][(mr + 16) * 32 + kb]);
        v16h b0 = load_frag(&Bl[cur][nc * 32 + kb]);
        v16h b1 = load_frag(&Bl[cur][(nc + 16) * 32 + kb]);
        acc00 = __builtin_amdgcn_wmma_f32_16x16x32_f16(false, a0, false, b0, (short)0, acc00, false, false);
        acc01 = __builtin_amdgcn_wmma_f32_16x16x32_f16(false, a0, false, b1, (short)0, acc01, false, false);
        acc10 = __builtin_amdgcn_wmma_f32_16x16x32_f16(false, a1, false, b0, (short)0, acc10, false, false);
        acc11 = __builtin_amdgcn_wmma_f32_16x16x32_f16(false, a1, false, b1, (short)0, acc11, false, false);
        __syncthreads();                // all reads done before buffer reuse
    }

    // D layout: VGPR i -> M=i (lanes 0-15) / M=8+i (lanes 16-31), N = lane&15.
    const int mb = g * Mg + m0 + wm * 32 + ((lane >> 4) << 3);
    const int nb = n0 + wn * 32 + (lane & 15);
#pragma unroll
    for (int i = 0; i < 8; ++i) {
        Y[(size_t)(mb + i) * N + nb]           = acc00[i] + bias[mb + i];
        Y[(size_t)(mb + i) * N + nb + 16]      = acc01[i] + bias[mb + i];
        Y[(size_t)(mb + 16 + i) * N + nb]      = acc10[i] + bias[mb + 16 + i];
        Y[(size_t)(mb + 16 + i) * N + nb + 16] = acc11[i] + bias[mb + 16 + i];
    }
}

// ---------------------------------------------------------------------------
// GroupNorm (+affine) + GELU for the value stack; writes f16 GEMM operand with
// contiguous-K columns (column n = b*L + l, K index = channel).
// src element (b,c,l) = src[b*sb + c*sc + l].  One block per (b, group).
// ---------------------------------------------------------------------------
__global__ __launch_bounds__(256)
void gn_gelu_value(const float* __restrict__ src, int sb, int sc,
                   const float* __restrict__ gw, const float* __restrict__ gb,
                   _Float16* __restrict__ Xo, int C, int L, int gs, int KX) {
    __shared__ float red[512];
    const int ngroups = C / gs;
    const int b   = blockIdx.x / ngroups;
    const int g   = blockIdx.x % ngroups;
    const int tid = threadIdx.x;
    const int cnt = gs * L;

    float s = 0.f, q = 0.f;
    for (int i = tid; i < cnt; i += 256) {
        const int c = g * gs + i / L, l = i % L;
        const float v = src[(size_t)b * sb + (size_t)c * sc + l];
        s += v; q += v * v;
    }
    red[tid] = s; red[256 + tid] = q;
    __syncthreads();
    for (int st = 128; st > 0; st >>= 1) {
        if (tid < st) { red[tid] += red[tid + st]; red[256 + tid] += red[256 + tid + st]; }
        __syncthreads();
    }
    const float mean = red[0] / cnt;
    const float var  = red[256] / cnt - mean * mean;
    const float rstd = rsqrtf(var + EPSF);

    for (int i = tid; i < cnt; i += 256) {
        const int c = g * gs + i / L, l = i % L;
        const float v = src[(size_t)b * sb + (size_t)c * sc + l];
        const float y = gelu_f((v - mean) * rstd * gw[c] + gb[c]);
        Xo[(size_t)(b * L + l) * KX + c] = (_Float16)y;
    }
}

// ---------------------------------------------------------------------------
// Gate stack input: xr = x.reshape(B*64, 8*256); GN over G=8 groups (fixed d,
// all l) + GELU; writes f16 columns of K=2048 (k = d*256 + l, column = sample).
// One block per (sample, d): grid 1024, 256 threads (= l).
// ---------------------------------------------------------------------------
__global__ __launch_bounds__(256)
void gn_gelu_gate_in(const float* __restrict__ x,
                     const float* __restrict__ gw, const float* __restrict__ gb,
                     _Float16* __restrict__ Xo) {
    __shared__ float red[512];
    const int bi = blockIdx.x;          // 0..1023
    const int smp = bi >> 3, d = bi & 7;
    const int b = smp >> 6, cg = smp & 63;
    const int tid = threadIdx.x;        // l
    const float v = x[(size_t)b * (512 * 256) + (size_t)(cg * 8 + d) * 256 + tid];
    red[tid] = v; red[256 + tid] = v * v;
    __syncthreads();
    for (int st = 128; st > 0; st >>= 1) {
        if (tid < st) { red[tid] += red[tid + st]; red[256 + tid] += red[256 + tid + st]; }
        __syncthreads();
    }
    const float mean = red[0] * (1.f / 256.f);
    const float var  = red[256] * (1.f / 256.f) - mean * mean;
    const float rstd = rsqrtf(var + EPSF);
    const int ch = d * 256 + tid;
    const float y = gelu_f((v - mean) * rstd * gw[ch] + gb[ch]);
    Xo[(size_t)smp * 2048 + ch] = (_Float16)y;
}

// Gate mid-stack GN: groups of 8 consecutive channels, length 1 (8 elems per
// group) -> one thread per (column n, group).  src Y is [M][128] f32.
__global__ __launch_bounds__(256)
void gn_gelu_gate_mid(const float* __restrict__ Y, int M,
                      const float* __restrict__ gw, const float* __restrict__ gb,
                      _Float16* __restrict__ Xo) {
    const int gt = blockIdx.x * 256 + threadIdx.x;
    if (gt >= (M >> 3) * 128) return;
    const int n = gt & 127, grp = gt >> 7;
    float v[8], s = 0.f, q = 0.f;
#pragma unroll
    for (int j = 0; j < 8; ++j) {
        v[j] = Y[(size_t)(grp * 8 + j) * 128 + n];
        s += v[j]; q += v[j] * v[j];
    }
    const float mean = s * 0.125f;
    const float rstd = rsqrtf(q * 0.125f - mean * mean + EPSF);
#pragma unroll
    for (int j = 0; j < 8; ++j) {
        const int c = grp * 8 + j;
        Xo[(size_t)n * M + c] = (_Float16)gelu_f((v[j] - mean) * rstd * gw[c] + gb[c]);
    }
}

// Final gate LayerNorm over all 256 outputs per sample, then exp().
__global__ __launch_bounds__(256)
void gate_ln_exp(const float* __restrict__ Y4,
                 const float* __restrict__ lnw, const float* __restrict__ lnb,
                 float* __restrict__ gmat) {
    __shared__ float red[512];
    const int n = blockIdx.x, c = threadIdx.x;
    const float v = Y4[(size_t)c * 128 + n];
    red[c] = v; red[256 + c] = v * v;
    __syncthreads();
    for (int st = 128; st > 0; st >>= 1) {
        if (c < st) { red[c] += red[c + st]; red[256 + c] += red[256 + c + st]; }
        __syncthreads();
    }
    const float mean = red[0] * (1.f / 256.f);
    const float rstd = rsqrtf(red[256] * (1.f / 256.f) - mean * mean + EPSF);
    gmat[(size_t)n * 256 + c] = expf((v - mean) * rstd * lnw[c] + lnb[c]);
}

// x_{t+1} = x_t + exp_gate[b,cg,l] * rz[cg] * v4[b,c1,l]   (ReZero combine)
__global__ __launch_bounds__(256)
void combine_step(const float* __restrict__ xold, const float* __restrict__ gmat,
                  const float* __restrict__ rz, const float* __restrict__ v4,
                  float* __restrict__ xnew) {
    const int idx = blockIdx.x * 256 + threadIdx.x;   // < 2*512*256
    const int b  = idx >> 17;
    const int r  = idx & 131071;
    const int c1 = r >> 8, l = r & 255;
    const int cg = c1 >> 3;
    const int smp = (b << 6) + cg;
    xnew[idx] = xold[idx] +
                gmat[(size_t)smp * 256 + l] * rz[cg] *
                v4[(size_t)c1 * 512 + b * 256 + l];
}

__global__ __launch_bounds__(256)
void cvt_f32_f16(const float* __restrict__ s, _Float16* __restrict__ d, int n) {
    const int i = blockIdx.x * 256 + threadIdx.x;
    if (i < n) d[i] = (_Float16)s[i];
}

// ---------------------------------------------------------------------------
// Host orchestration
// ---------------------------------------------------------------------------
enum {
    IN_X = 0,
    IN_G_GN1_W, IN_G_GN1_B, IN_G_C1_W, IN_G_C1_B,
    IN_G_GN2_W, IN_G_GN2_B, IN_G_C2_W, IN_G_C2_B,
    IN_G_GN3_W, IN_G_GN3_B, IN_G_C3_W, IN_G_C3_B,
    IN_G_GN4_W, IN_G_GN4_B, IN_G_C4_W, IN_G_C4_B,
    IN_G_LN_W, IN_G_LN_B,
    IN_V_GN1_W, IN_V_GN1_B, IN_V_C1_W, IN_V_C1_B,
    IN_V_GN2_W, IN_V_GN2_B, IN_V_C2_W, IN_V_C2_B,
    IN_V_GN3_W, IN_V_GN3_B, IN_V_C3_W, IN_V_C3_B,
    IN_V_GN4_W, IN_V_GN4_B, IN_V_C4_W, IN_V_C4_B,
    IN_V_RZ
};

extern "C" void kernel_launch(void* const* d_in, const int* in_sizes, int n_in,
                              void* d_out, int out_size, void* d_ws, size_t ws_size,
                              hipStream_t stream) {
    (void)in_sizes; (void)n_in; (void)out_size; (void)ws_size;

    const float* F[40];
    for (int i = 0; i < 36; ++i) F[i] = (const float*)d_in[i];

    // Workspace carve-out (all 256B aligned; total ~13.5 MB).
    char* base = (char*)d_ws;
    size_t off = 0;
    auto alloc = [&](size_t bytes) -> char* {
        char* p = base + off;
        off = (off + bytes + 255) & ~(size_t)255;
        return p;
    };
    float*     xA   = (float*)alloc(262144 * 4);
    float*     xB   = (float*)alloc(262144 * 4);
    float*     Yv   = (float*)alloc(1024 * 512 * 4);   // value GEMM outputs (reused)
    float*     Yg   = (float*)alloc(1024 * 128 * 4);   // gate GEMM outputs (reused)
    _Float16*  Xc   = (_Float16*)alloc(512 * 1024 * 2);  // value f16 operand [n][K]
    _Float16*  Xg   = (_Float16*)alloc(128 * 2048 * 2);  // gate  f16 operand [n][K]
    float*     gmat = (float*)alloc(128 * 256 * 4);
    _Float16*  wv1  = (_Float16*)alloc((size_t)1024 * 512 * 2);
    _Float16*  wv2  = (_Float16*)alloc((size_t)1024 * 64 * 2);
    _Float16*  wv3  = (_Float16*)alloc((size_t)1024 * 64 * 2);
    _Float16*  wv4  = (_Float16*)alloc((size_t)512 * 1024 * 2);
    _Float16*  wg1  = (_Float16*)alloc((size_t)1024 * 2048 * 2);
    _Float16*  wg2  = (_Float16*)alloc((size_t)1024 * 64 * 2);
    _Float16*  wg3  = (_Float16*)alloc((size_t)1024 * 64 * 2);
    _Float16*  wg4  = (_Float16*)alloc((size_t)256 * 1024 * 2);

    hipMemcpyAsync(xA, F[IN_X], 262144 * sizeof(float), hipMemcpyDeviceToDevice, stream);

    auto cvt = [&](const float* s, _Float16* d, int n) {
        cvt_f32_f16<<<(n + 255) / 256, 256, 0, stream>>>(s, d, n);
    };
    cvt(F[IN_V_C1_W], wv1, 1024 * 512);
    cvt(F[IN_V_C2_W], wv2, 1024 * 64);
    cvt(F[IN_V_C3_W], wv3, 1024 * 64);
    cvt(F[IN_V_C4_W], wv4, 512 * 1024);
    cvt(F[IN_G_C1_W], wg1, 1024 * 2048);
    cvt(F[IN_G_C2_W], wg2, 1024 * 64);
    cvt(F[IN_G_C3_W], wg3, 1024 * 64);
    cvt(F[IN_G_C4_W], wg4, 256 * 1024);

    for (int t = 0; t < 64; ++t) {
        const float* xcur = (t & 1) ? xB : xA;
        float* xnext = (t == 63) ? (float*)d_out : ((t & 1) ? xA : xB);

        // ---- value net (N = 512 columns = (b,l)) ----
        gn_gelu_value<<<128, 256, 0, stream>>>(xcur, 512 * 256, 256,
                                               F[IN_V_GN1_W], F[IN_V_GN1_B],
                                               Xc, 512, 256, 8, 512);
        gemm_f16_wmma<<<dim3(16, 4, 1), 256, 0, stream>>>(wv1, Xc, F[IN_V_C1_B], Yv,
                                                          1024, 512, 512, 512);
        gn_gelu_value<<<256, 256, 0, stream>>>(Yv, 256, 512,
                                               F[IN_V_GN2_W], F[IN_V_GN2_B],
                                               Xc, 1024, 256, 8, 1024);
        gemm_f16_wmma<<<dim3(1, 4, 16), 256, 0, stream>>>(wv2, Xc, F[IN_V_C2_B], Yv,
                                                          64, 512, 64, 1024);
        gn_gelu_value<<<256, 256, 0, stream>>>(Yv, 256, 512,
                                               F[IN_V_GN3_W], F[IN_V_GN3_B],
                                               Xc, 1024, 256, 8, 1024);
        gemm_f16_wmma<<<dim3(1, 4, 16), 256, 0, stream>>>(wv3, Xc, F[IN_V_C3_B], Yv,
                                                          64, 512, 64, 1024);
        gn_gelu_value<<<256, 256, 0, stream>>>(Yv, 256, 512,
                                               F[IN_V_GN4_W], F[IN_V_GN4_B],
                                               Xc, 1024, 256, 8, 1024);
        gemm_f16_wmma<<<dim3(8, 4, 1), 256, 0, stream>>>(wv4, Xc, F[IN_V_C4_B], Yv,
                                                         512, 512, 1024, 1024);

        // ---- gate net (N = 128 columns = samples) ----
        gn_gelu_gate_in<<<1024, 256, 0, stream>>>(xcur, F[IN_G_GN1_W], F[IN_G_GN1_B], Xg);
        gemm_f16_wmma<<<dim3(16, 1, 1), 256, 0, stream>>>(wg1, Xg, F[IN_G_C1_B], Yg,
                                                          1024, 128, 2048, 2048);
        gn_gelu_gate_mid<<<64, 256, 0, stream>>>(Yg, 1024, F[IN_G_GN2_W], F[IN_G_GN2_B], Xg);
        gemm_f16_wmma<<<dim3(1, 1, 16), 256, 0, stream>>>(wg2, Xg, F[IN_G_C2_B], Yg,
                                                          64, 128, 64, 1024);
        gn_gelu_gate_mid<<<64, 256, 0, stream>>>(Yg, 1024, F[IN_G_GN3_W], F[IN_G_GN3_B], Xg);
        gemm_f16_wmma<<<dim3(1, 1, 16), 256, 0, stream>>>(wg3, Xg, F[IN_G_C3_B], Yg,
                                                          64, 128, 64, 1024);
        gn_gelu_gate_mid<<<64, 256, 0, stream>>>(Yg, 1024, F[IN_G_GN4_W], F[IN_G_GN4_B], Xg);
        gemm_f16_wmma<<<dim3(4, 1, 1), 256, 0, stream>>>(wg4, Xg, F[IN_G_C4_B], Yg,
                                                         256, 128, 1024, 1024);
        gate_ln_exp<<<128, 256, 0, stream>>>(Yg, F[IN_G_LN_W], F[IN_G_LN_B], gmat);

        // ---- combine ----
        combine_step<<<1024, 256, 0, stream>>>(xcur, gmat, F[IN_V_RZ], Yv, xnext);
    }
}
